// LocalDecoder_1211180777860
// MI455X (gfx1250) — matrix operand order repaired and checked by reading, not compiled, not run on hardware
//
#include <hip/hip_runtime.h>

// ---------------------------------------------------------------------------
// Types for CDNA5 WMMA (gfx1250, wave32)
// ---------------------------------------------------------------------------
typedef __attribute__((ext_vector_type(16))) __bf16 v16bf;
typedef __attribute__((ext_vector_type(2)))  __bf16 v2bf;
typedef __attribute__((ext_vector_type(8)))  float  v8f;

union FragU {
  v16bf        v;
  unsigned int u[8];
  uint4        q[2];
};

// Packed f32x2 -> bf16x2 (matches v_cvt_pk_bf16_f32)
__device__ __forceinline__ unsigned int pack2bf(float a, float b) {
  v2bf p;
  p.x = (__bf16)a;
  p.y = (__bf16)b;
  return __builtin_bit_cast(unsigned int, p);
}

// ---------------------------------------------------------------------------
// Tiled GEMM: C[M,N] = A[M,K](f32) * W[K,N](f32) + bias[N], via bf16 WMMA.
// Block: 256 threads (8 waves), tile 128x128, K-step 32, software-pipelined:
// next K-tile is loaded to registers while WMMA consumes the current LDS tile
// (s_wait_loadcnt sinks past the 8 WMMAs -> global latency hidden).
// Waves: 4 (M) x 2 (N); each wave: 2x4 fragments of 16x16.
// Requires: M % 128 == 0, N % 128 == 0, K % 32 == 0 (true for all call sites).
// NTSTORE: non-temporal C stores (streaming output, keep W resident in L2).
// ---------------------------------------------------------------------------
template <bool NTSTORE>
__global__ __launch_bounds__(256) void gemm_bf16(
    const float* __restrict__ A, int lda,
    const float* __restrict__ Wg, int N,          // ldw == N
    const float* __restrict__ bias,
    float* __restrict__ C, int ldc,
    int M, int K)
{
  __shared__ __bf16 As[128 * 32];   // [row][k]
  __shared__ __bf16 Bs[128 * 32];   // [n][k]  (transposed weight tile)

  const int tid  = threadIdx.x;
  const int wave = tid >> 5;
  const int lane = tid & 31;
  const int bm   = blockIdx.y * 128;
  const int bn   = blockIdx.x * 128;
  const int wm   = (wave >> 1) * 32;   // wave row offset in tile
  const int wn   = (wave & 1) * 64;    // wave col offset in tile
  const int ml   = lane & 15;
  const int hl   = lane >> 4;

  v8f acc[2][4] = {};
  float4 fa[4], fw[4];

  // ---- prologue: load + stage K-tile 0 ------------------------------------
  #pragma unroll
  for (int i = 0; i < 4; ++i) {
    const int q = tid + i * 256;               // 0..1023 float4s
    fa[i] = *(const float4*)(A  + (long)(bm + (q >> 3)) * lda + ((q & 7) << 2));
    fw[i] = *(const float4*)(Wg + (long)(q >> 5) * N + bn + ((q & 31) << 2));
  }
  #pragma unroll
  for (int i = 0; i < 4; ++i) {
    const int q  = tid + i * 256;
    *(uint2*)(&As[(q >> 3) * 32 + ((q & 7) << 2)]) =
        make_uint2(pack2bf(fa[i].x, fa[i].y), pack2bf(fa[i].z, fa[i].w));
    const int kk = q >> 5, n4 = (q & 31) << 2;
    Bs[(n4 + 0) * 32 + kk] = (__bf16)fw[i].x;
    Bs[(n4 + 1) * 32 + kk] = (__bf16)fw[i].y;
    Bs[(n4 + 2) * 32 + kk] = (__bf16)fw[i].z;
    Bs[(n4 + 3) * 32 + kk] = (__bf16)fw[i].w;
  }
  __syncthreads();

  for (int k0 = 0; k0 < K; k0 += 32) {
    const bool more = (k0 + 32) < K;

    // ---- issue next tile's global loads (consumed after the barrier) ------
    if (more) {
      #pragma unroll
      for (int i = 0; i < 4; ++i) {
        const int q = tid + i * 256;
        fa[i] = *(const float4*)(A  + (long)(bm + (q >> 3)) * lda + k0 + 32 + ((q & 7) << 2));
        fw[i] = *(const float4*)(Wg + (long)(k0 + 32 + (q >> 5)) * N + bn + ((q & 31) << 2));
      }
    }

    // ---- fragments + WMMA on current LDS tile ------------------------------
    FragU a[2], b[4];
    #pragma unroll
    for (int m = 0; m < 2; ++m) {
      // ISA 16-bit A layout: half 0 -> K {0..7,16..23}; half 1 -> K {8..15,24..31}
      const __bf16* ap = &As[(wm + m * 16 + ml) * 32 + hl * 8];
      a[m].q[0] = *(const uint4*)(ap);
      a[m].q[1] = *(const uint4*)(ap + 16);
    }
    #pragma unroll
    for (int n = 0; n < 4; ++n) {
      // 16-bit B layout (matches A): half 0 -> K {0..7,16..23}; half 1 -> K {8..15,24..31}
      const __bf16* bp = &Bs[(wn + n * 16 + ml) * 32 + hl * 8];
      b[n].q[0] = *(const uint4*)(bp);
      b[n].q[1] = *(const uint4*)(bp + 16);
    }
    #pragma unroll
    for (int m = 0; m < 2; ++m)
      #pragma unroll
      for (int n = 0; n < 4; ++n)
        acc[m][n] = __builtin_amdgcn_wmma_f32_16x16x32_bf16(
            false, a[m].v, false, b[n].v, (short)0, acc[m][n], false, false);
    __syncthreads();

    // ---- stage next tile into LDS -----------------------------------------
    if (more) {
      #pragma unroll
      for (int i = 0; i < 4; ++i) {
        const int q  = tid + i * 256;
        *(uint2*)(&As[(q >> 3) * 32 + ((q & 7) << 2)]) =
            make_uint2(pack2bf(fa[i].x, fa[i].y), pack2bf(fa[i].z, fa[i].w));
        const int kk = q >> 5, n4 = (q & 31) << 2;
        Bs[(n4 + 0) * 32 + kk] = (__bf16)fw[i].x;
        Bs[(n4 + 1) * 32 + kk] = (__bf16)fw[i].y;
        Bs[(n4 + 2) * 32 + kk] = (__bf16)fw[i].z;
        Bs[(n4 + 3) * 32 + kk] = (__bf16)fw[i].w;
      }
      __syncthreads();
    }
  }

  // ---- epilogue: + bias, store f32 (M,N multiples of 128: no guards) ------
  #pragma unroll
  for (int m = 0; m < 2; ++m)
    #pragma unroll
    for (int n = 0; n < 4; ++n) {
      const int col = bn + wn + n * 16 + ml;
      const float bv = bias[col];
      #pragma unroll
      for (int r = 0; r < 8; ++r) {
        const int row = bm + wm + m * 16 + hl * 8 + r;
        const float val = acc[m][n][r] + bv;
        float* dst = C + (long)row * ldc + col;
        if (NTSTORE) __builtin_nontemporal_store(val, dst);
        else         *dst = val;
      }
    }
}

// ---------------------------------------------------------------------------
// Non-overlapping windowed MHSA. One block per (window, head, batch).
// 256 threads = 8 waves; wave w owns query rows w*16..w*16+15.
// QK^T and PV on v_wmma_f32_16x16x32_bf16; softmax in registers with
// __shfl_xor over the 16-lane half that holds each row.
// qkv layout: [B, T, 3*1024] (q | k | v), dh = 128, H = 8, W = 128.
// ---------------------------------------------------------------------------
__global__ __launch_bounds__(256) void attn_win(
    const float* __restrict__ qkv, float* __restrict__ out, int T)
{
  __shared__ __bf16 vst[128 * 128];      // V transposed: [d][key]      (32 KB)
  __shared__ __bf16 ps[8][16 * 128];     // per-wave P (bf16)           (32 KB)

  const int tid  = threadIdx.x;
  const int wave = tid >> 5;
  const int lane = tid & 31;
  const int ml   = lane & 15;
  const int hl   = lane >> 4;
  const int win  = blockIdx.x;
  const int h    = blockIdx.y;
  const int b    = blockIdx.z;

  const long base  = ((long)b * T + (long)win * 128) * 3072;
  const float* qb = qkv + base + (long)h * 128;
  const float* kb = qkv + base + 1024 + (long)h * 128;
  const float* vb = qkv + base + 2048 + (long)h * 128;

  // ---- stage V transposed into LDS as bf16 --------------------------------
  #pragma unroll 4
  for (int i = 0; i < 64; ++i) {
    int idx = tid + i * 256;
    int key = idx >> 7;
    int d   = idx & 127;
    vst[d * 128 + key] = (__bf16)vb[(long)key * 3072 + d];
  }

  const int m0 = wave * 16;

  // ---- S = Q K^T (8 col-fragments of 16 per wave, K-loop over dh) ---------
  v8f sacc[8] = {};
  for (int kc = 0; kc < 128; kc += 32) {
    FragU af;
    {
      const float* qp = qb + (long)(m0 + ml) * 3072 + kc + hl * 8;
      float4 f0 = *(const float4*)(qp);
      float4 f1 = *(const float4*)(qp + 4);
      float4 f2 = *(const float4*)(qp + 16);
      float4 f3 = *(const float4*)(qp + 20);
      af.u[0] = pack2bf(f0.x, f0.y); af.u[1] = pack2bf(f0.z, f0.w);
      af.u[2] = pack2bf(f1.x, f1.y); af.u[3] = pack2bf(f1.z, f1.w);
      af.u[4] = pack2bf(f2.x, f2.y); af.u[5] = pack2bf(f2.z, f2.w);
      af.u[6] = pack2bf(f3.x, f3.y); af.u[7] = pack2bf(f3.z, f3.w);
    }
    #pragma unroll
    for (int nf = 0; nf < 8; ++nf) {
      FragU bf_;
      const float* kp = kb + (long)(nf * 16 + ml) * 3072 + kc + hl * 8;
      float4 g0 = *(const float4*)(kp);
      float4 g1 = *(const float4*)(kp + 4);
      float4 g2 = *(const float4*)(kp + 16);
      float4 g3 = *(const float4*)(kp + 20);
      bf_.u[0] = pack2bf(g0.x, g0.y); bf_.u[1] = pack2bf(g0.z, g0.w);
      bf_.u[2] = pack2bf(g1.x, g1.y); bf_.u[3] = pack2bf(g1.z, g1.w);
      bf_.u[4] = pack2bf(g2.x, g2.y); bf_.u[5] = pack2bf(g2.z, g2.w);
      bf_.u[6] = pack2bf(g3.x, g3.y); bf_.u[7] = pack2bf(g3.z, g3.w);
      sacc[nf] = __builtin_amdgcn_wmma_f32_16x16x32_bf16(
          false, af.v, false, bf_.v, (short)0, sacc[nf], false, false);
    }
  }

  // ---- softmax over each row (128 values: 8 frags x 16 lanes in half) -----
  const float scale = 0.0883883476483184f;  // 1/sqrt(128)
  #pragma unroll
  for (int r = 0; r < 8; ++r) {
    float mx = -3.0e38f;
    #pragma unroll
    for (int nf = 0; nf < 8; ++nf) {
      sacc[nf][r] *= scale;
      mx = fmaxf(mx, sacc[nf][r]);
    }
    #pragma unroll
    for (int off = 1; off < 16; off <<= 1)
      mx = fmaxf(mx, __shfl_xor(mx, off, 32));
    float sum = 0.f;
    #pragma unroll
    for (int nf = 0; nf < 8; ++nf) {
      float e = __expf(sacc[nf][r] - mx);
      sacc[nf][r] = e;
      sum += e;
    }
    #pragma unroll
    for (int off = 1; off < 16; off <<= 1)
      sum += __shfl_xor(sum, off, 32);
    const float inv = 1.0f / sum;
    #pragma unroll
    for (int nf = 0; nf < 8; ++nf)
      ps[wave][(hl * 8 + r) * 128 + nf * 16 + ml] = (__bf16)(sacc[nf][r] * inv);
  }
  __syncthreads();   // vst staged by all waves; ps visible for re-fragmenting

  // ---- O = P V  (A from ps, B from vst) -----------------------------------
  v8f oacc[8] = {};
  const __bf16* pw = &ps[wave][0];
  for (int kc = 0; kc < 128; kc += 32) {
    FragU af;
    const __bf16* ap = pw + ml * 128 + kc + hl * 8;
    af.q[0] = *(const uint4*)(ap);
    af.q[1] = *(const uint4*)(ap + 16);
    #pragma unroll
    for (int nf = 0; nf < 8; ++nf) {
      FragU bf_;
      const __bf16* vp = &vst[(nf * 16 + ml) * 128 + kc + hl * 8];
      bf_.q[0] = *(const uint4*)(vp);
      bf_.q[1] = *(const uint4*)(vp + 16);
      oacc[nf] = __builtin_amdgcn_wmma_f32_16x16x32_bf16(
          false, af.v, false, bf_.v, (short)0, oacc[nf], false, false);
    }
  }

  // ---- store attention output (f32, [B,T,1024]) ---------------------------
  const long ob = ((long)b * T + (long)win * 128) * 1024 + (long)h * 128;
  #pragma unroll
  for (int nf = 0; nf < 8; ++nf)
    #pragma unroll
    for (int r = 0; r < 8; ++r)
      out[ob + (long)(m0 + hl * 8 + r) * 1024 + nf * 16 + ml] = oacc[nf][r];
}

// ---------------------------------------------------------------------------
// LayerNorm over D=1024 (optionally with residual add). One block per row.
// ---------------------------------------------------------------------------
__global__ __launch_bounds__(256) void ln_rows(
    const float* __restrict__ x, const float* __restrict__ res,
    const float* __restrict__ g, const float* __restrict__ bet,
    float* __restrict__ out)
{
  __shared__ float red[8];
  const int row = blockIdx.x, tid = threadIdx.x;
  const int lane = tid & 31, wave = tid >> 5;
  const float* xr = x + (long)row * 1024;
  const float* rr = res ? res + (long)row * 1024 : nullptr;

  float v[4];
  float s = 0.f;
  #pragma unroll
  for (int i = 0; i < 4; ++i) {
    int d = tid + i * 256;
    float t = xr[d];
    if (rr) t += rr[d];
    v[i] = t; s += t;
  }
  #pragma unroll
  for (int off = 1; off < 32; off <<= 1) s += __shfl_xor(s, off, 32);
  if (lane == 0) red[wave] = s;
  __syncthreads();
  float tot = 0.f;
  #pragma unroll
  for (int w = 0; w < 8; ++w) tot += red[w];
  const float mu = tot * (1.0f / 1024.0f);
  __syncthreads();

  float d2 = 0.f;
  #pragma unroll
  for (int i = 0; i < 4; ++i) { v[i] -= mu; d2 += v[i] * v[i]; }
  #pragma unroll
  for (int off = 1; off < 32; off <<= 1) d2 += __shfl_xor(d2, off, 32);
  if (lane == 0) red[wave] = d2;
  __syncthreads();
  float vt = 0.f;
  #pragma unroll
  for (int w = 0; w < 8; ++w) vt += red[w];
  const float inv = rsqrtf(vt * (1.0f / 1024.0f) + 1e-5f);
  #pragma unroll
  for (int i = 0; i < 4; ++i) {
    int d = tid + i * 256;
    out[(long)row * 1024 + d] = v[i] * inv * g[d] + bet[d];
  }
}

// ---------------------------------------------------------------------------
// Unmerge: counts + gather/normalize
// ---------------------------------------------------------------------------
__global__ void zero_f32(float* __restrict__ p, int n) {
  int i = blockIdx.x * blockDim.x + threadIdx.x;
  if (i < n) p[i] = 0.f;
}

__global__ void unmerge_count(const int* __restrict__ src,
                              float* __restrict__ counts, int L, int Mm) {
  int i = blockIdx.x * blockDim.x + threadIdx.x;   // over B*L (B=2)
  if (i >= 2 * L) return;
  int b = i / L;
  atomicAdd(&counts[b * Mm + src[i]], 1.0f);
}

__global__ void unmerge_gather(const float* __restrict__ xin,
                               const int* __restrict__ src,
                               const float* __restrict__ counts,
                               float* __restrict__ xout, int L, int Mm) {
  long i = (long)blockIdx.x * blockDim.x + threadIdx.x;   // over B*L*256 float4
  if (i >= (long)2 * L * 256) return;
  int  b   = (int)(i / ((long)L * 256));
  long rem = i - (long)b * L * 256;
  int  pos = (int)(rem >> 8);
  int  c4  = (int)(rem & 255);
  int  s   = src[b * L + pos];
  float inv = 1.0f / fmaxf(counts[b * Mm + s], 1.0f);
  const float4 f = *(const float4*)(xin + ((long)b * Mm + s) * 1024 + c4 * 4);
  float4 o;
  o.x = f.x * inv; o.y = f.y * inv; o.z = f.z * inv; o.w = f.w * inv;
  *(float4*)(xout + ((long)b * L + pos) * 1024 + c4 * 4) = o;
}

// ---------------------------------------------------------------------------
// Orchestration.  B=2, D=1024, H=8, V=32000, L0=2048, M1=1024, M2=512, W=128.
// ---------------------------------------------------------------------------
extern "C" void kernel_launch(void* const* d_in, const int* in_sizes, int n_in,
                              void* d_out, int out_size, void* d_ws, size_t ws_size,
                              hipStream_t stream) {
  (void)in_sizes; (void)n_in; (void)out_size; (void)ws_size;

  const float* merged = (const float*)d_in[0];
  const int*   src0   = (const int*)d_in[1];
  const int*   src1   = (const int*)d_in[2];
  const float* qkv_w0 = (const float*)d_in[3];
  const float* qkv_b0 = (const float*)d_in[4];
  const float* o_w0   = (const float*)d_in[5];
  const float* o_b0   = (const float*)d_in[6];
  const float* ln_g0  = (const float*)d_in[7];
  const float* ln_b0  = (const float*)d_in[8];
  const float* qkv_w1 = (const float*)d_in[9];
  const float* qkv_b1 = (const float*)d_in[10];
  const float* o_w1   = (const float*)d_in[11];
  const float* o_b1   = (const float*)d_in[12];
  const float* ln_g1  = (const float*)d_in[13];
  const float* ln_b1  = (const float*)d_in[14];
  const float* fln_g  = (const float*)d_in[15];
  const float* fln_b  = (const float*)d_in[16];
  const float* out_w  = (const float*)d_in[17];
  const float* out_b  = (const float*)d_in[18];
  float* logits = (float*)d_out;

  char* ws = (char*)d_ws;
  const size_t MB = 1ull << 20;
  float* actA   = (float*)(ws + 0);        // 16 MB  (B*L0*D)
  float* actB   = (float*)(ws + 16 * MB);  // 16 MB
  float* qkvb   = (float*)(ws + 32 * MB);  // 48 MB  (B*L0*3D) — reused for o-proj out
  float* attnb  = (float*)(ws + 80 * MB);  // 16 MB
  float* counts = (float*)(ws + 96 * MB);  //  8 KB

  const dim3 T256(256);

  // ===== layer 1 path (T = M1 = 1024) =====
  zero_f32<<<dim3(4), T256, 0, stream>>>(counts, 1024);                 // 2*512
  unmerge_count<<<dim3(8), T256, 0, stream>>>(src1, counts, 1024, 512);
  unmerge_gather<<<dim3(2048), T256, 0, stream>>>(merged, src1, counts, actA, 1024, 512);

  gemm_bf16<false><<<dim3(24, 16), T256, 0, stream>>>(actA, 1024, qkv_w1, 3072, qkv_b1,
                                                      qkvb, 3072, 2048, 1024);
  attn_win<<<dim3(8, 8, 2), T256, 0, stream>>>(qkvb, attnb, 1024);
  gemm_bf16<false><<<dim3(8, 16), T256, 0, stream>>>(attnb, 1024, o_w1, 1024, o_b1,
                                                     qkvb, 1024, 2048, 1024);
  ln_rows<<<dim3(2048), T256, 0, stream>>>(actA, qkvb, ln_g1, ln_b1, actB);

  // ===== unmerge to L0 = 2048 =====
  zero_f32<<<dim3(8), T256, 0, stream>>>(counts, 2048);                 // 2*1024
  unmerge_count<<<dim3(16), T256, 0, stream>>>(src0, counts, 2048, 1024);
  unmerge_gather<<<dim3(4096), T256, 0, stream>>>(actB, src0, counts, actA, 2048, 1024);

  // ===== layer 0 path (T = L0 = 2048) =====
  gemm_bf16<false><<<dim3(24, 32), T256, 0, stream>>>(actA, 1024, qkv_w0, 3072, qkv_b0,
                                                      qkvb, 3072, 4096, 1024);
  attn_win<<<dim3(16, 8, 2), T256, 0, stream>>>(qkvb, attnb, 2048);
  gemm_bf16<false><<<dim3(8, 32), T256, 0, stream>>>(attnb, 1024, o_w0, 1024, o_b0,
                                                     qkvb, 1024, 4096, 1024);
  ln_rows<<<dim3(4096), T256, 0, stream>>>(actA, qkvb, ln_g0, ln_b0, actB);

  // ===== final LN + vocab projection (streaming NT stores for logits) =====
  ln_rows<<<dim3(4096), T256, 0, stream>>>(actB, nullptr, fln_g, fln_b, actA);
  gemm_bf16<true><<<dim3(250, 32), T256, 0, stream>>>(actA, 1024, out_w, 32000, out_b,
                                                      logits, 32000, 4096, 1024);
}
